// SimVQ_5781025980421
// MI455X (gfx1250) — matrix operand order repaired and checked by reading, not compiled, not run on hardware
//
#include <hip/hip_runtime.h>
#include <hip/hip_bf16.h>

typedef __attribute__((ext_vector_type(8)))  __bf16 v8bf;
typedef __attribute__((ext_vector_type(16))) __bf16 v16bf;
typedef __attribute__((ext_vector_type(8)))  float  v8f;

#define E_DIM 512
#define N_E   8192
#define HW    1024          // 32*32
#define NB    16            // batch
#define NTOK  16384         // 16*32*32
#define NELEM 8388608       // NTOK * E_DIM

// k_dist blocking
#define MT    4             // row-tiles per block (64 rows)
#define SC    32            // codes staged in LDS per iteration
#define RS    (E_DIM + 16)  // padded LDS row stride in ushorts (+32B: 8-bank shift/row)
#define NSC   (N_E / SC)    // 256 staged iterations

static __device__ __forceinline__ unsigned short f32_to_bf16_bits(float f) {
    union { float f; unsigned int u; } v; v.f = f;
    unsigned int lsb = (v.u >> 16) & 1u;
    return (unsigned short)((v.u + 0x7fffu + lsb) >> 16);
}

// ---- WMMA fragment loaders (bf16 stored as ushort), contiguous k order (k = 16h + s), same for A and B ----
// A (16x32, MxK): lanes 0-15 hold M=lane, K=0..15; lanes 16-31 K=16..31
static __device__ __forceinline__ v16bf load_A(const unsigned short* __restrict__ base,
                                               int lda, int lane) {
    int m  = lane & 15;
    int kb = (lane < 16) ? 0 : 16;
    const unsigned short* p = base + (size_t)m * lda + kb;
    v8bf lo = *(const v8bf*)(p);
    v8bf hi = *(const v8bf*)(p + 8);
    return __builtin_shufflevector(lo, hi, 0,1,2,3,4,5,6,7,8,9,10,11,12,13,14,15);
}
// B (32x16, KxN) with B[k,n] = mem[n*ldb + k] (row-major [N,K] source => implicit transpose):
// lanes 0-15: N=lane, K=0..15 ; lanes 16-31: N=lane-16, K=16..31
static __device__ __forceinline__ v16bf load_B(const unsigned short* __restrict__ base,
                                               int ldb, int lane) {
    int n  = lane & 15;
    int kb = (lane < 16) ? 0 : 16;
    return *(const v16bf*)(base + (size_t)n * ldb + kb);
}

// ---------------- fp32 -> bf16 elementwise ----------------
__global__ void k_f2bf(const float* __restrict__ in, unsigned short* __restrict__ out, int n) {
    int i = blockIdx.x * blockDim.x + threadIdx.x;
    if (i < n) out[i] = f32_to_bf16_bits(in[i]);
}

// ---------------- z [B,C,H,W] -> z_bf16 [B*HW, C] (LDS-tiled transpose) ----------------
__global__ void k_zt(const float* __restrict__ z, unsigned short* __restrict__ zb) {
    __shared__ float tile[32][33];
    int tx = threadIdx.x, ty = threadIdx.y;
    int hw0 = blockIdx.x * 32, c0 = blockIdx.y * 32, b = blockIdx.z;
#pragma unroll
    for (int j = 0; j < 4; j++) {
        int c = c0 + ty + j * 8;
        tile[ty + j * 8][tx] = z[((size_t)(b * E_DIM + c)) * HW + hw0 + tx];
    }
    __syncthreads();
#pragma unroll
    for (int j = 0; j < 4; j++) {
        int hw = hw0 + ty + j * 8;
        zb[((size_t)(b * HW + hw)) * E_DIM + c0 + tx] = f32_to_bf16_bits(tile[tx][ty + j * 8]);
    }
}

// ------- qc = emb @ proj_w^T + b  (bf16 WMMA, fp32 accum, dual K-chains) -------
__global__ void __launch_bounds__(256)
k_proj(const unsigned short* __restrict__ embB, const unsigned short* __restrict__ pwB,
       const float* __restrict__ bias, float* __restrict__ qc, unsigned short* __restrict__ qcB) {
    int wave = (blockIdx.x * blockDim.x + threadIdx.x) >> 5;
    int lane = threadIdx.x & 31;
    int tn = wave & 31;   // 512/16 N-tiles
    int tm = wave >> 5;   // 8192/16 M-tiles
    const unsigned short* arow = embB + (size_t)tm * 16 * E_DIM;
    const unsigned short* brow = pwB  + (size_t)tn * 16 * E_DIM;  // B[k,n]=proj_w[n,k]
    v8f acc0 = {0.f,0.f,0.f,0.f,0.f,0.f,0.f,0.f};
    v8f acc1 = {0.f,0.f,0.f,0.f,0.f,0.f,0.f,0.f};
#pragma unroll
    for (int k0 = 0; k0 < E_DIM; k0 += 64) {     // two independent WMMA chains
        v16bf a0 = load_A(arow + k0,      E_DIM, lane);
        v16bf b0 = load_B(brow + k0,      E_DIM, lane);
        v16bf a1 = load_A(arow + k0 + 32, E_DIM, lane);
        v16bf b1 = load_B(brow + k0 + 32, E_DIM, lane);
        acc0 = __builtin_amdgcn_wmma_f32_16x16x32_bf16(false, a0, false, b0,
                                                       (short)0, acc0, false, false);
        acc1 = __builtin_amdgcn_wmma_f32_16x16x32_bf16(false, a1, false, b1,
                                                       (short)0, acc1, false, false);
    }
    int n  = tn * 16 + (lane & 15);
    float bn = bias[n];
    int mrow = tm * 16 + ((lane >> 4) << 3);   // D layout: M = v + 8*(lane/16)
#pragma unroll
    for (int v = 0; v < 8; v++) {
        float val = acc0[v] + acc1[v] + bn;
        size_t o = (size_t)(mrow + v) * E_DIM + n;
        qc[o]  = val;
        qcB[o] = f32_to_bf16_bits(val);
    }
}

// ---------------- per-row ||qc||^2, one wave per row ----------------
__global__ void k_qcn(const float* __restrict__ qc, float* __restrict__ qcn) {
    int row  = (blockIdx.x * blockDim.x + threadIdx.x) >> 5;
    int lane = threadIdx.x & 31;
    const float* r = qc + (size_t)row * E_DIM;
    float s = 0.f;
#pragma unroll 4
    for (int j = lane; j < E_DIM; j += 32) { float x = r[j]; s += x * x; }
#pragma unroll
    for (int off = 16; off; off >>= 1) s += __shfl_xor(s, off, 32);
    if (lane == 0) qcn[row] = s;
}

// ------- distance GEMM (bf16 WMMA) fused with argmin over 8192 codes -------
// Block = 64 z-rows (4 row-tiles) x all codes. Codes staged through a
// double-buffered LDS tile of 32 codes so one L2 read of qc serves 64 rows
// (4x less L2 traffic than M-tile=16). 8 waves = 4 row-tiles x 2 code-stripes.
// Per-lane deferred argmin; one butterfly + LDS combine at the end.
__global__ void __launch_bounds__(256)
k_dist(const unsigned short* __restrict__ zB, const unsigned short* __restrict__ qcB,
       const float* __restrict__ qcn, int* __restrict__ idx, float* __restrict__ idxf) {
    __shared__ __align__(32) unsigned short sB[2][SC * RS];   // 2 x 33 KB staged qc
    __shared__ float s_s[8][16];
    __shared__ int   s_i[8][16];
    int tid  = threadIdx.x;
    int wid  = tid >> 5, lane = tid & 31;
    int rt   = wid >> 1;                 // row-tile within block (0..3)
    int half = wid & 1;                  // which 16 of the staged 32 codes

    const unsigned short* arow = zB + ((size_t)blockIdx.x * MT + rt) * 16 * E_DIM;
    v16bf a[16];                          // K=512 of this wave's 16 z-rows
#pragma unroll
    for (int kk = 0; kk < 16; kk++) a[kk] = load_A(arow + kk * 32, E_DIM, lane);

    float best[8]; int bidx[8];
#pragma unroll
    for (int v = 0; v < 8; v++) { best[v] = 3.4e38f; bidx[v] = 0; }

    // staging: wave wid copies rows {wid, wid+8, wid+16, wid+24}; 32 lanes x 2 uint4 per row
    uint4 r[8];
    auto g_load = [&](int c0) {
#pragma unroll
        for (int rr = 0; rr < 4; rr++)
#pragma unroll
            for (int j = 0; j < 2; j++) {
                int row = wid + rr * 8;
                int u   = lane + j * 32;                  // uint4 index within row
                r[rr * 2 + j] =
                    *(const uint4*)(qcB + ((size_t)(c0 + row)) * E_DIM + (size_t)u * 8);
            }
    };
    auto s_store = [&](int buf) {
#pragma unroll
        for (int rr = 0; rr < 4; rr++)
#pragma unroll
            for (int j = 0; j < 2; j++) {
                int row = wid + rr * 8;
                int u   = lane + j * 32;
                *(uint4*)(&sB[buf][(size_t)row * RS + (size_t)u * 8]) = r[rr * 2 + j];
            }
    };

    g_load(0);
    for (int sc = 0; sc < NSC; sc++) {               // 256 uniform iterations
        int buf = sc & 1;
        s_store(buf);
        __syncthreads();
        if (sc + 1 < NSC) g_load((sc + 1) * SC);     // overlaps the WMMAs below

        const unsigned short* brow = &sB[buf][(size_t)(half * 16) * RS];
        const unsigned short* bl   = brow + (size_t)(lane & 15) * RS + ((lane < 16) ? 0 : 16);
        v8f acc0 = {0.f,0.f,0.f,0.f,0.f,0.f,0.f,0.f};
        v8f acc1 = {0.f,0.f,0.f,0.f,0.f,0.f,0.f,0.f};
#pragma unroll
        for (int kk = 0; kk < 8; kk++) {             // two independent WMMA chains
            v16bf b0 = *(const v16bf*)(bl + kk * 32);
            v16bf b1 = *(const v16bf*)(bl + (kk + 8) * 32);
            acc0 = __builtin_amdgcn_wmma_f32_16x16x32_bf16(false, a[kk],     false, b0,
                                                           (short)0, acc0, false, false);
            acc1 = __builtin_amdgcn_wmma_f32_16x16x32_bf16(false, a[kk + 8], false, b1,
                                                           (short)0, acc1, false, false);
        }
        int n  = sc * SC + half * 16 + (lane & 15);
        float qn = qcn[n];
#pragma unroll
        for (int v = 0; v < 8; v++) {
            float s = qn - 2.0f * (acc0[v] + acc1[v]);
            if (s < best[v]) { best[v] = s; bidx[v] = n; }   // strict < keeps first index
        }
        __syncthreads();
    }

    // one 16-lane butterfly per output row (half-waves reduce independently)
#pragma unroll
    for (int v = 0; v < 8; v++) {
        float s = best[v]; int ci = bidx[v];
#pragma unroll
        for (int off = 1; off < 16; off <<= 1) {
            float so = __shfl_xor(s, off, 32);
            int   co = __shfl_xor(ci, off, 32);
            if (so < s || (so == s && co < ci)) { s = so; ci = co; }
        }
        best[v] = s; bidx[v] = ci;
    }
    if ((lane & 15) == 0) {
        int h8 = (lane >> 4) * 8;
#pragma unroll
        for (int v = 0; v < 8; v++) { s_s[wid][h8 + v] = best[v];
                                      s_i[wid][h8 + v] = bidx[v]; }
    }
    __syncthreads();
    if (tid < 64) {                       // 64 rows per block
        int rt2 = tid >> 4, rl = tid & 15;
        int w0 = rt2 * 2;
        float bs = s_s[w0][rl];     int bi = s_i[w0][rl];
        float s2 = s_s[w0 + 1][rl]; int i2 = s_i[w0 + 1][rl];
        if (s2 < bs || (s2 == bs && i2 < bi)) { bs = s2; bi = i2; }
        int row = blockIdx.x * (MT * 16) + tid;
        idx[row]  = bi;
        idxf[row] = (float)bi;
    }
}

// ------- gather qc[idx] -> z_q [B,C,H,W]; fused commit-loss reduction -------
__global__ void k_out(const float* __restrict__ z, const float* __restrict__ qc,
                      const int* __restrict__ idx, float* __restrict__ zq,
                      float* __restrict__ loss) {
    __shared__ float red[256];
    size_t i = (size_t)blockIdx.x * 256 + threadIdx.x;   // index in [B,C,HW] layout
    int hw = (int)(i & (HW - 1));
    int bc = (int)(i >> 10);
    int c  = bc & (E_DIM - 1);
    int b  = bc >> 9;
    int n  = b * HW + hw;
    float zv = z[i];
    float q  = qc[(size_t)idx[n] * E_DIM + c];
    zq[i] = q;                                  // straight-through forward == z_q
    float d = q - zv;
    red[threadIdx.x] = d * d;
    __syncthreads();
#pragma unroll
    for (int s = 128; s; s >>= 1) {
        if (threadIdx.x < s) red[threadIdx.x] += red[threadIdx.x + s];
        __syncthreads();
    }
    if (threadIdx.x == 0)
        atomicAdd(loss, red[0] * (1.25f / (float)NELEM));  // (BETA + 1) * mean
}

extern "C" void kernel_launch(void* const* d_in, const int* in_sizes, int n_in,
                              void* d_out, int out_size, void* d_ws, size_t ws_size,
                              hipStream_t stream) {
    const float* z   = (const float*)d_in[0];
    const float* emb = (const float*)d_in[1];
    const float* pw  = (const float*)d_in[2];
    const float* pb  = (const float*)d_in[3];

    float* out_zq   = (float*)d_out;
    float* out_loss = out_zq + NELEM;
    float* out_idx  = out_loss + 1;

    // workspace carve-up (256B aligned)
    char* ws = (char*)d_ws; size_t o = 0;
    auto carve = [&](size_t bytes) { void* p = ws + o; o = (o + bytes + 255) & ~(size_t)255; return p; };
    unsigned short* embB = (unsigned short*)carve((size_t)N_E  * E_DIM * 2);
    unsigned short* pwB  = (unsigned short*)carve((size_t)E_DIM* E_DIM * 2);
    unsigned short* zBuf = (unsigned short*)carve((size_t)NTOK * E_DIM * 2);
    float*          qcF  = (float*)         carve((size_t)N_E  * E_DIM * 4);
    unsigned short* qcB  = (unsigned short*)carve((size_t)N_E  * E_DIM * 2);
    float*          qcn  = (float*)         carve((size_t)N_E  * 4);
    int*            idxI = (int*)           carve((size_t)NTOK * 4);

    hipMemsetAsync(out_loss, 0, sizeof(float), stream);

    k_f2bf<<<(N_E * E_DIM + 255) / 256, 256, 0, stream>>>(emb, embB, N_E * E_DIM);
    k_f2bf<<<(E_DIM * E_DIM + 255) / 256, 256, 0, stream>>>(pw, pwB, E_DIM * E_DIM);

    dim3 tb(32, 8), tg(HW / 32, E_DIM / 32, NB);
    k_zt<<<tg, tb, 0, stream>>>(z, zBuf);

    // 8192/16 * 512/16 = 16384 wave-tiles, 8 waves per 256-thread block
    k_proj<<<2048, 256, 0, stream>>>(embB, pwB, pb, qcF, qcB);
    k_qcn<<<N_E / 8, 256, 0, stream>>>(qcF, qcn);
    k_dist<<<NTOK / (MT * 16), 256, 0, stream>>>(zBuf, qcB, qcn, idxI, out_idx);
    k_out<<<NELEM / 256, 256, 0, stream>>>(z, qcF, idxI, out_zq, out_loss);
}
